// Model_56693568307444
// MI455X (gfx1250) — matrix operand order repaired and checked by reading, not compiled, not run on hardware
//
#include <hip/hip_runtime.h>

// ---------------------------------------------------------------------------
// Problem constants (from reference): B=16, T=16384, R=8, K=2, W=32
// Conv stack dilations 1,2,4,8,16; total left pad = 31.
// Padded time axis: buffer index tt == padded-x index; layer l computes
// out[tt] = f(in[tt], in[tt + d_l]); final h lives at tt in [0,T).
// ---------------------------------------------------------------------------
#define B_   16
#define T_   16384
#define R_   8
#define W_   32
#define TP_  16416   // computed time range for layers 0..3 (1026 tiles of 16)
#define TPA_ 16432   // allocated time extent (covers tt+dil reads, zero tail)
#define NT_FULL 1026
#define NT_LAST 1024
#define SLOPE_ 0.22916666666666666f   // (1/8 + 1/3)/2

// LDS staging: 16 rows x 64 ch, row stride padded to 72 halfs (144B) so the
// 16 column-lanes of a B-fragment read map to distinct LDS banks.
#define ROWS_ 72
#define SLAB_ (16 * ROWS_)

typedef __attribute__((ext_vector_type(16))) _Float16 v16h;
typedef __attribute__((ext_vector_type(8)))  _Float16 v8h;
typedef __attribute__((ext_vector_type(8)))  float    v8f;
typedef __attribute__((ext_vector_type(4)))  int      i4;

#if defined(__HIP_DEVICE_COMPILE__) && \
    __has_builtin(__builtin_amdgcn_global_load_async_to_lds_b128) && \
    __has_builtin(__builtin_amdgcn_s_wait_asynccnt)
#define HAVE_ASYNC_LDS 1
#else
#define HAVE_ASYNC_LDS 0
#endif

__device__ __forceinline__ float rrelu_f(float y) {
  return (y >= 0.0f) ? y : y * SLOPE_;
}

// Assemble a 16-half B fragment from two 16B-aligned LDS reads.
__device__ __forceinline__ v16h ld_frag16(const _Float16* p) {
  v8h lo = *(const v8h*)p;
  v8h hi = *(const v8h*)(p + 8);
  return __builtin_shufflevector(lo, hi, 0, 1, 2, 3, 4, 5, 6, 7,
                                 8, 9, 10, 11, 12, 13, 14, 15);
}

// ---------------------------------------------------------------------------
// Zero-init the two f16 ping-pong activation buffers (tails must be 0).
// ---------------------------------------------------------------------------
__global__ void k_zero(uint4* __restrict__ p, long n) {
  long i = (long)blockIdx.x * blockDim.x + threadIdx.x;
  if (i < n) p[i] = make_uint4(0u, 0u, 0u, 0u);
}

// ---------------------------------------------------------------------------
// Pre-pack conv weights (layers 1..4) into WMMA A-fragment layout (f16).
// A is 16x32 (MxK), contiguous k order used here (B matches):
//   lane: m = lane&15, hi = lane>>4 ; element e: k = hi*16 + e
// k_global = q*32 + k ; tap = k_global>=64 ; ci = k_global & 63.
// 13 groups: layer1 cot0..3, layer2 cot0..3, layer3 cot0..3, layer4 cot0.
// Group stride = 4 chunks * 32 lanes * 16 halfs = 2048 halfs.
// ---------------------------------------------------------------------------
__global__ void k_prep(const float* __restrict__ cw1, const float* __restrict__ cw2,
                       const float* __restrict__ cw3, const float* __restrict__ cw4,
                       _Float16* __restrict__ wprep) {
  int g    = blockIdx.x;          // 0..12
  int q    = threadIdx.x >> 5;    // chunk 0..3
  int lane = threadIdx.x & 31;
  int layer = (g < 4) ? 1 : (g < 8) ? 2 : (g < 12) ? 3 : 4;
  int cot   = (layer == 4) ? 0 : (g & 3);
  const float* cw = (layer == 1) ? cw1 : (layer == 2) ? cw2 : (layer == 3) ? cw3 : cw4;
  int coMax = (layer == 4) ? 8 : 64;
  int m  = lane & 15;
  int hi = lane >> 4;
  int co = cot * 16 + m;
  _Float16* dst = wprep + (size_t)g * 2048 + (size_t)(q * 32 + lane) * 16;
  for (int e = 0; e < 16; ++e) {
    int kl  = hi * 16 + e;
    int kg  = q * 32 + kl;
    int tap = kg >> 6;
    int ci  = kg & 63;
    float v = (co < coMax) ? cw[(co * 64 + ci) * 2 + tap] : 0.0f;
    dst[e] = (_Float16)v;
  }
}

// ---------------------------------------------------------------------------
// Layer 0: conv 1->64, k=2, dil=1, with implicit left pad of 31 zeros.
// Writes time-major f16 activations [b][tt][64].
// ---------------------------------------------------------------------------
__global__ void k_layer0(const float* __restrict__ x, const float* __restrict__ cw0,
                         const float* __restrict__ cb0, _Float16* __restrict__ out) {
  __shared__ float sw[128];
  __shared__ float sb[64];
  int tid = threadIdx.x;
  if (tid < 128) sw[tid] = cw0[tid];
  if (tid < 64)  sb[tid] = cb0[tid];
  __syncthreads();
  long idx = (long)blockIdx.x * blockDim.x + tid;
  if (idx >= (long)B_ * TP_) return;
  int b  = (int)(idx / TP_);
  int tt = (int)(idx % TP_);
  const float* xb = x + (size_t)b * T_;
  float x0 = (tt >= 31 && tt < 31 + T_) ? xb[tt - 31] : 0.0f;
  int t1 = tt + 1;
  float x1 = (t1 >= 31 && t1 < 31 + T_) ? xb[t1 - 31] : 0.0f;
  _Float16* o = out + ((size_t)b * TPA_ + tt) * 64;
  for (int co = 0; co < 64; ++co) {
    float y = fmaf(sw[co * 2], x0, fmaf(sw[co * 2 + 1], x1, sb[co]));
    o[co] = (_Float16)rrelu_f(y);
  }
}

// ---------------------------------------------------------------------------
// LDS-staged WMMA dilated conv 64->64 k=2 (layers 1..3).
// Block = 128 threads = 4 waves; wave w owns co-tile w. The 16-timestep
// input slab (both taps) is staged into LDS once per block via async
// Global->LDS DMA (ASYNCcnt) with double buffering, then all 4 waves pull
// B fragments from LDS (ds_load_b128, conflict-free via ROWS_=72 padding).
// B fragment layout: lane n = lane&15, hi = lane>>4, element e holds
// B[k = hi*16 + e][n] (16 contiguous channels in the time-major slab).
// ---------------------------------------------------------------------------
__global__ void k_conv_lds(const _Float16* __restrict__ in,
                           _Float16* __restrict__ out16,
                           const _Float16* __restrict__ afrag,
                           const float* __restrict__ bias,
                           int dil, int nTiles) {
  __shared__ _Float16 sIn[2][2 * SLAB_];   // [buf][tap*SLAB_ + row*ROWS_ + ch]
  int tid  = threadIdx.x;                  // 0..127
  int lane = tid & 31;
  int cot  = tid >> 5;                     // wave id == co-tile
  int b    = blockIdx.z;
  int n  = lane & 15;
  int hi = lane >> 4;

  v16h a0 = *(const v16h*)(afrag + ((size_t)(cot * 4 + 0) * 32 + lane) * 16);
  v16h a1 = *(const v16h*)(afrag + ((size_t)(cot * 4 + 1) * 32 + lane) * 16);
  v16h a2 = *(const v16h*)(afrag + ((size_t)(cot * 4 + 2) * 32 + lane) * 16);
  v16h a3 = *(const v16h*)(afrag + ((size_t)(cot * 4 + 3) * 32 + lane) * 16);

  const _Float16* inb = in + (size_t)b * TPA_ * 64;
  int row = tid >> 3;    // 0..15
  int seg = tid & 7;     // 8 x 16B segments per 128B row

  // Stage one tile (both taps) into buffer `buf`: 2 x b128 per thread.
  auto stage = [&](int tile, int buf) {
    int t0 = tile * 16;
    const _Float16* g0 = inb + (size_t)(t0 + row) * 64 + seg * 8;
    _Float16* d0 = &sIn[buf][row * ROWS_ + seg * 8];
#if HAVE_ASYNC_LDS
    __builtin_amdgcn_global_load_async_to_lds_b128(
        (__attribute__((address_space(1))) i4*)g0,
        (__attribute__((address_space(3))) i4*)d0, 0, 0);
    __builtin_amdgcn_global_load_async_to_lds_b128(
        (__attribute__((address_space(1))) i4*)(g0 + (size_t)dil * 64),
        (__attribute__((address_space(3))) i4*)(d0 + SLAB_), 0, 0);
#else
    *(v8h*)d0           = *(const v8h*)g0;
    *(v8h*)(d0 + SLAB_) = *(const v8h*)(g0 + (size_t)dil * 64);
#endif
  };

  int tile = blockIdx.x;
  if (tile >= nTiles) return;          // uniform per block
  stage(tile, 0);
  int buf = 0;
  for (; tile < nTiles; tile += gridDim.x) {
    int nxt = tile + gridDim.x;
    bool hasNext = nxt < nTiles;
    if (hasNext) stage(nxt, buf ^ 1);  // overlap DMA with this tile's math
#if HAVE_ASYNC_LDS
    if (hasNext) __builtin_amdgcn_s_wait_asynccnt(2);
    else         __builtin_amdgcn_s_wait_asynccnt(0);
#endif
    __syncthreads();                   // current buffer visible to all waves

    const _Float16* s = &sIn[buf][0];
    int off0 = n * ROWS_ + hi * 16;    // tap0, k-chunk 0..31
    v16h b0 = ld_frag16(s + off0);
    v16h b1 = ld_frag16(s + off0 + 32);
    v16h b2 = ld_frag16(s + SLAB_ + off0);
    v16h b3 = ld_frag16(s + SLAB_ + off0 + 32);

    v8f c = {};
    c = __builtin_amdgcn_wmma_f32_16x16x32_f16(false, a0, false, b0, (short)0, c, false, false);
    c = __builtin_amdgcn_wmma_f32_16x16x32_f16(false, a1, false, b1, (short)0, c, false, false);
    c = __builtin_amdgcn_wmma_f32_16x16x32_f16(false, a2, false, b2, (short)0, c, false, false);
    c = __builtin_amdgcn_wmma_f32_16x16x32_f16(false, a3, false, b3, (short)0, c, false, false);

    int co0 = cot * 16 + hi * 8;       // lane holds rows co0..co0+7 at column n
    int tA  = tile * 16 + n;
    v8h hout;
#pragma unroll
    for (int v = 0; v < 8; ++v) {
      float y = c[v] + bias[co0 + v];
      hout[v] = (_Float16)rrelu_f(y);
    }
    *(v8h*)(out16 + ((size_t)b * TPA_ + tA) * 64 + co0) = hout;

    __syncthreads();                   // done reading buf before re-staging it
    buf ^= 1;
  }
}

// ---------------------------------------------------------------------------
// Direct WMMA dilated conv (used for layer 4: 64 -> 8, zero-padded A,
// f32 output into hbuf[b][t][16]). One wave per block, A frags hoisted.
// ---------------------------------------------------------------------------
template<bool F32OUT>
__global__ void k_conv_wmma(const _Float16* __restrict__ in,
                            _Float16* __restrict__ out16,
                            float* __restrict__ out32,
                            const _Float16* __restrict__ afrag,
                            const float* __restrict__ bias, int nCo,
                            int dil, int nTiles) {
  int lane = threadIdx.x;          // blockDim.x == 32
  int cot  = blockIdx.y;
  int b    = blockIdx.z;
  int n  = lane & 15;
  int hi = lane >> 4;

  v16h a0 = *(const v16h*)(afrag + ((size_t)(cot * 4 + 0) * 32 + lane) * 16);
  v16h a1 = *(const v16h*)(afrag + ((size_t)(cot * 4 + 1) * 32 + lane) * 16);
  v16h a2 = *(const v16h*)(afrag + ((size_t)(cot * 4 + 2) * 32 + lane) * 16);
  v16h a3 = *(const v16h*)(afrag + ((size_t)(cot * 4 + 3) * 32 + lane) * 16);

  const _Float16* inb = in + (size_t)b * TPA_ * 64;

  for (int tile = blockIdx.x; tile < nTiles; tile += gridDim.x) {
    int t0 = tile * 16;
    int tA = t0 + n;
    int tB = tA + dil;
    const _Float16* pA = inb + (size_t)tA * 64 + hi * 16;
    const _Float16* pB = inb + (size_t)tB * 64 + hi * 16;
    v16h b0 = *(const v16h*)(pA);
    v16h b1 = *(const v16h*)(pA + 32);
    v16h b2 = *(const v16h*)(pB);
    v16h b3 = *(const v16h*)(pB + 32);

    int nxt = tile + gridDim.x;
    if (nxt < nTiles)
      __builtin_prefetch(inb + ((size_t)(nxt * 16 + n)) * 64 + hi * 16, 0, 1);

    v8f c = {};
    c = __builtin_amdgcn_wmma_f32_16x16x32_f16(false, a0, false, b0, (short)0, c, false, false);
    c = __builtin_amdgcn_wmma_f32_16x16x32_f16(false, a1, false, b1, (short)0, c, false, false);
    c = __builtin_amdgcn_wmma_f32_16x16x32_f16(false, a2, false, b2, (short)0, c, false, false);
    c = __builtin_amdgcn_wmma_f32_16x16x32_f16(false, a3, false, b3, (short)0, c, false, false);

    int co0 = cot * 16 + hi * 8;
    if (F32OUT) {
      float* o = out32 + ((size_t)b * T_ + tA) * 16 + co0;   // cot == 0 here
#pragma unroll
      for (int v = 0; v < 8; ++v) {
        int co  = co0 + v;
        float y = c[v] + ((co < nCo) ? bias[co] : 0.0f);
        o[v] = rrelu_f(y);
      }
    } else {
      v8h h;
#pragma unroll
      for (int v = 0; v < 8; ++v) {
        float y = c[v] + bias[co0 + v];
        h[v] = (_Float16)rrelu_f(y);
      }
      *(v8h*)(out16 + ((size_t)b * TPA_ + tA) * 64 + co0) = h;
    }
  }
}

// ---------------------------------------------------------------------------
// Pointwise: gumbel-softmax (temp 0.1) over R=8 -> gen_z, then the three
// 1x1 "m" convs (16->16->16->1) -> mus. h channels live in hbuf[b][t][0..7].
// ---------------------------------------------------------------------------
__global__ void k_point(const float* __restrict__ hbuf, const float* __restrict__ gumbel,
                        const float* __restrict__ mw0, const float* __restrict__ mb0,
                        const float* __restrict__ mw1, const float* __restrict__ mb1,
                        const float* __restrict__ mw2, const float* __restrict__ mb2,
                        float* __restrict__ genz, float* __restrict__ mus) {
  __shared__ float s0[256], s1[256], sb0[16], sb1[16], s2[16];
  __shared__ float sb2;
  int tid = threadIdx.x;
  if (tid < 256) { s0[tid] = mw0[tid]; s1[tid] = mw1[tid]; }
  if (tid < 16)  { sb0[tid] = mb0[tid]; sb1[tid] = mb1[tid]; s2[tid] = mw2[tid]; }
  if (tid == 0)  sb2 = mb2[0];
  __syncthreads();

  long idx = (long)blockIdx.x * blockDim.x + tid;
  if (idx >= (long)B_ * T_) return;
  int b = (int)(idx / T_);
  int t = (int)(idx % T_);

  const float* h8 = hbuf + idx * 16;
  const float* g8 = gumbel + idx * 8;

  float v[16];
  float li[8];
  float mx = -1e30f;
#pragma unroll
  for (int r = 0; r < 8; ++r) {
    float hv = h8[r];
    v[8 + r] = hv;                       // concat([gen_z, h]) channels 8..15
    li[r] = (hv + g8[r]) * 10.0f;        // / TEMP
    mx = fmaxf(mx, li[r]);
  }
  float s = 0.0f;
#pragma unroll
  for (int r = 0; r < 8; ++r) { li[r] = __expf(li[r] - mx); s += li[r]; }
  float inv = 1.0f / s;
#pragma unroll
  for (int r = 0; r < 8; ++r) {
    float z = li[r] * inv;
    v[r] = z;                            // channels 0..7
    genz[((size_t)b * 8 + r) * T_ + t] = z;
  }

  float m0[16];
#pragma unroll
  for (int o = 0; o < 16; ++o) {
    float acc = sb0[o];
#pragma unroll
    for (int i = 0; i < 16; ++i) acc = fmaf(s0[o * 16 + i], v[i], acc);
    m0[o] = rrelu_f(acc);
  }
  float m1[16];
#pragma unroll
  for (int o = 0; o < 16; ++o) {
    float acc = sb1[o];
#pragma unroll
    for (int i = 0; i < 16; ++i) acc = fmaf(s1[o * 16 + i], m0[i], acc);
    m1[o] = rrelu_f(acc);
  }
  float mu = sb2;
#pragma unroll
  for (int i = 0; i < 16; ++i) mu = fmaf(s2[i], m1[i], mu);
  mus[idx] = mu;
}

// ---------------------------------------------------------------------------
// Final windowed correction:
// _mus[b,t'] = mus[b,W+t'] - cov(b,W+t') * sum_w fj(b,w,W+t') *
//              (x[b,w+t'] - mus[b,w+t'])
// cov = (sum_r z_r * sqrt_cov_tt[r])^2 ; fj = sum_r z_r * fish_jt[r,w]
// ---------------------------------------------------------------------------
__global__ void k_final(const float* __restrict__ x, const float* __restrict__ genz,
                        const float* __restrict__ mus, const float* __restrict__ sc,
                        const float* __restrict__ fish, float* __restrict__ outmus) {
  __shared__ float sf[R_ * W_];
  __shared__ float ss[R_];
  int tid = threadIdx.x;
  if (tid < R_ * W_) sf[tid] = fish[tid];
  if (tid < R_)      ss[tid] = sc[tid];
  __syncthreads();

  long idx = (long)blockIdx.x * blockDim.x + tid;
  long total = (long)B_ * (T_ - W_);
  if (idx >= total) return;
  int b  = (int)(idx / (T_ - W_));
  int tp = (int)(idx % (T_ - W_));
  int t  = W_ + tp;

  float z[8];
  float cv = 0.0f;
#pragma unroll
  for (int r = 0; r < 8; ++r) {
    z[r] = genz[((size_t)b * 8 + r) * T_ + t];
    cv = fmaf(z[r], ss[r], cv);
  }
  cv = cv * cv;

  const float* xb = x   + (size_t)b * T_;
  const float* mb = mus + (size_t)b * T_;
  float acc = 0.0f;
  for (int w = 0; w < W_; ++w) {
    float fj = 0.0f;
#pragma unroll
    for (int r = 0; r < 8; ++r) fj = fmaf(z[r], sf[r * W_ + w], fj);
    acc = fmaf(fj, xb[w + tp] - mb[w + tp], acc);
  }
  outmus[idx] = mb[t] - cv * acc;
}

// ---------------------------------------------------------------------------
// Host launcher
// ---------------------------------------------------------------------------
extern "C" void kernel_launch(void* const* d_in, const int* in_sizes, int n_in,
                              void* d_out, int out_size, void* d_ws, size_t ws_size,
                              hipStream_t stream) {
  (void)in_sizes; (void)n_in; (void)out_size; (void)ws_size;

  const float* x    = (const float*)d_in[0];
  const float* cw0  = (const float*)d_in[1];
  const float* cb0  = (const float*)d_in[2];
  const float* cw1  = (const float*)d_in[3];
  const float* cb1  = (const float*)d_in[4];
  const float* cw2  = (const float*)d_in[5];
  const float* cb2  = (const float*)d_in[6];
  const float* cw3  = (const float*)d_in[7];
  const float* cb3  = (const float*)d_in[8];
  const float* cw4  = (const float*)d_in[9];
  const float* cb4  = (const float*)d_in[10];
  const float* mw0  = (const float*)d_in[11];
  const float* mb0  = (const float*)d_in[12];
  const float* mw1  = (const float*)d_in[13];
  const float* mb1  = (const float*)d_in[14];
  const float* mw2  = (const float*)d_in[15];
  const float* mb2  = (const float*)d_in[16];
  const float* sct  = (const float*)d_in[17];   // sqrt_cov_tt (R,1)
  const float* fish = (const float*)d_in[18];   // fish_jt (R,W)
  const float* gum  = (const float*)d_in[19];   // gumbel (B,T,R)

  // Workspace layout
  char* ws = (char*)d_ws;
  const size_t SZ_ACT = (size_t)B_ * TPA_ * 64 * sizeof(_Float16);   // ~33.7 MB
  _Float16* bufA  = (_Float16*)(ws);
  _Float16* bufB  = (_Float16*)(ws + SZ_ACT);
  float*    hbuf  = (float*)(ws + 2 * SZ_ACT);                       // (B,T,16) f32
  const size_t SZ_H = (size_t)B_ * T_ * 16 * sizeof(float);
  float*    musb  = (float*)(ws + 2 * SZ_ACT + SZ_H);                // (B,T) f32
  const size_t SZ_MUS = (size_t)B_ * T_ * sizeof(float);
  _Float16* wprep = (_Float16*)(ws + 2 * SZ_ACT + SZ_H + SZ_MUS);    // 13*2048 halfs

  float* out_mus  = (float*)d_out;                                   // (B, T-W)
  float* out_genz = out_mus + (size_t)B_ * (T_ - W_);                // (B, R, T)

  // 1) zero the ping-pong activation buffers (valid-tail zeros)
  long zeroN = (long)((2 * SZ_ACT) / sizeof(uint4));
  k_zero<<<dim3((unsigned)((zeroN + 255) / 256)), 256, 0, stream>>>((uint4*)ws, zeroN);

  // 2) pack WMMA A-fragments for layers 1..4
  k_prep<<<13, 128, 0, stream>>>(cw1, cw2, cw3, cw4, wprep);

  // 3) layer 0 (1->64, dil 1) into bufA (time-major f16)
  long n0 = (long)B_ * TP_;
  k_layer0<<<dim3((unsigned)((n0 + 255) / 256)), 256, 0, stream>>>(x, cw0, cb0, bufA);

  // 4) WMMA conv stack: layers 1-3 LDS-staged (async DMA, 4 waves/block),
  //    layer 4 direct with f32 epilogue.
  dim3 gFull(96, 1, B_);
  k_conv_lds<<<gFull, 128, 0, stream>>>(bufA, bufB, wprep,          cb1, 2, NT_FULL);
  k_conv_lds<<<gFull, 128, 0, stream>>>(bufB, bufA, wprep +  8192,  cb2, 4, NT_FULL);
  k_conv_lds<<<gFull, 128, 0, stream>>>(bufA, bufB, wprep + 16384,  cb3, 8, NT_FULL);
  dim3 gLast(128, 1, B_);
  k_conv_wmma<true><<<gLast, 32, 0, stream>>>(bufB, nullptr, hbuf, wprep + 24576,
                                              cb4, 8, 16, NT_LAST);

  // 5) softmax + m-convs -> gen_z (into d_out) and mus
  long np = (long)B_ * T_;
  k_point<<<dim3((unsigned)((np + 255) / 256)), 256, 0, stream>>>(
      hbuf, gum, mw0, mb0, mw1, mb1, mw2, mb2, out_genz, musb);

  // 6) final windowed correction -> _mus (into d_out)
  long nf = (long)B_ * (T_ - W_);
  k_final<<<dim3((unsigned)((nf + 255) / 256)), 256, 0, stream>>>(
      x, out_genz, musb, sct, fish, out_mus);
}
